// GraphNameEncoder_41214506172688
// MI455X (gfx1250) — compile-verified, matrix operand order checked by reading, not run on hardware
//
#include <hip/hip_runtime.h>

// ---------------------------------------------------------------------------
// GraphNameEncoder for MI455X (gfx1250, wave32, WMMA).
// Shapes: N=8192, L=16, F=256, D=768, H=4, DH=192, 3D=2304.
//
// v2 changes vs v1:
//  * 4 samples / block (M=64, 4 WMMA M-tiles): halves per-block weight
//    re-reads from L2 (2048 blocks x 4.7MB = ~9.6GB), making the kernel
//    compute-bound on the bf16 WMMA pipe instead of L2-bound.
//  * 4 M-tile accumulators interleaved in every K-chunk: one B fragment is
//    reused by 4 back-to-back v_wmma, so each s_wait_dscnt covers 4 WMMAs.
//  * Attention processed head-by-head so Q/K/V slices (3 x 24KB) fit LDS
//    alongside persistent emb (96KB) and ctx (96KB). Total LDS ~= 270KB.
//  * V stored transposed with stride 16 (no zero padding in LDS); the
//    K=16->32 pad lives in registers (upper fragment half zeroed).
// ---------------------------------------------------------------------------

typedef __attribute__((ext_vector_type(16))) __bf16 v16bf;
typedef __attribute__((ext_vector_type(8)))  __bf16 v8bf;
typedef __attribute__((ext_vector_type(4)))  __bf16 v4bf;
typedef __attribute__((ext_vector_type(8)))  float  v8f;

#define D_     768
#define TD_    2304          // 3*D
#define DH_    192           // head dim
#define OUTW_  1024          // F + D
#define NW1_   (TD_ * D_)    // in_proj_w elements
#define NW2_   (D_ * D_)     // out_proj_w elements

// LDS carve (bytes). Total = 276,480 (< 320 KB per WGP).
#define EMB_OFF  0                        // bf16 [64][768]  embeddings
#define CTX_OFF  98304                    // bf16 [64][768]  attention output
#define QH_OFF   196608                   // bf16 [64][192]  Q, current head
#define KH_OFF   221184                   // bf16 [64][192]  K, current head
#define VTH_OFF  245760                   // bf16 [4][192][16] V^T, current head
#define SC_OFF   270336                   // f32  [4][16][16] scores
#define AA_OFF   274432                   // bf16 [4][16][16] attn weights
#define SMEM_BYTES 276480

// Load a 16-bit A/B-style WMMA fragment from a row-major matrix.
// ISA 7.12.2: lane m = lane&15 selects the row; lanes 0-15 hold
// K = {0..7, 16..23}, lanes 16-31 hold K = {8..15, 24..31}: two contiguous
// 8-element (16-byte) runs per lane.
__device__ __forceinline__ v16bf ld_frag(const __bf16* base, int row0,
                                         int stride, int kbase) {
  const int lane = threadIdx.x & 31;
  const int m    = lane & 15;
  const int k0   = (lane & 16) ? 8 : 0;
  const __bf16* p = base + (row0 + m) * stride + kbase + k0;
  union { v16bf v; v8bf h[2]; } u;
  u.h[0] = *(const v8bf*)(p);
  u.h[1] = *(const v8bf*)(p + 16);
  return u.v;
}

// Fragment with only K=0..15 real data (K=16..31 zeroed in registers).
__device__ __forceinline__ v16bf ld_frag_k16(const __bf16* base, int row0,
                                             int stride) {
  const int lane = threadIdx.x & 31;
  const int m    = lane & 15;
  const int k0   = (lane & 16) ? 8 : 0;
  union { v16bf v; v8bf h[2]; } u;
  u.h[0] = *(const v8bf*)(base + (row0 + m) * stride + k0);
  v8bf z;
  #pragma unroll
  for (int i = 0; i < 8; ++i) z[i] = (__bf16)0.f;
  u.h[1] = z;
  return u.v;
}

__device__ __forceinline__ v8f wmma_bf16(v16bf a, v16bf b, v8f c) {
  // (neg_a, A, neg_b, B, c_mod, C, reuse_a, reuse_b)
  return __builtin_amdgcn_wmma_f32_16x16x32_bf16(false, a, false, b,
                                                 (short)0, c, false, false);
}

// ---------------------------------------------------------------------------
// Kernel 1: fp32 -> bf16 weight conversion into workspace.
// 2304 blocks x 256 threads x 4 elems = 2,359,296 = NW1_ + NW2_ exactly.
// ---------------------------------------------------------------------------
__global__ __launch_bounds__(256) void cvt_weights_kernel(
    const float* __restrict__ w1f, const float* __restrict__ w2f,
    unsigned short* __restrict__ w1u, unsigned short* __restrict__ w2u) {
  __bf16* o1 = (__bf16*)w1u;
  __bf16* o2 = (__bf16*)w2u;
  const int i = (blockIdx.x * 256 + threadIdx.x) * 4;
  const float* src;
  __bf16* dst;
  if (i < NW1_) { src = w1f + i; dst = o1 + i; }
  else          { src = w2f + (i - NW1_); dst = o2 + (i - NW1_); }
  float4 f = *(const float4*)src;
  v4bf b;
  b[0] = (__bf16)f.x; b[1] = (__bf16)f.y; b[2] = (__bf16)f.z; b[3] = (__bf16)f.w;
  *(v4bf*)dst = b;
}

// ---------------------------------------------------------------------------
// Kernel 2: fused encoder. 1 block = 4 samples, 256 threads = 8 waves.
// ---------------------------------------------------------------------------
__global__ __launch_bounds__(256) void encoder_kernel(
    const float* __restrict__ x, const int* __restrict__ names,
    const float* __restrict__ wte,
    const unsigned short* __restrict__ w1u, const float* __restrict__ b1,
    const unsigned short* __restrict__ w2u, const float* __restrict__ b2,
    float* __restrict__ out) {
  extern __shared__ char smem[];
  __bf16* emb = (__bf16*)(smem + EMB_OFF);
  __bf16* ctx = (__bf16*)(smem + CTX_OFF);
  __bf16* Qh  = (__bf16*)(smem + QH_OFF);
  __bf16* Kh  = (__bf16*)(smem + KH_OFF);
  __bf16* Vth = (__bf16*)(smem + VTH_OFF);
  float*  sc  = (float*) (smem + SC_OFF);
  __bf16* aa  = (__bf16*)(smem + AA_OFF);

  const __bf16* w1 = (const __bf16*)w1u;
  const __bf16* w2 = (const __bf16*)w2u;

  const int tid  = threadIdx.x;
  const int lane = tid & 31;
  const int wid  = tid >> 5;
  const int n0   = blockIdx.x * 4;           // first sample of this block
  const int mrow = (lane >> 4) * 8;          // C-tile row base for this lane

  // ---- Phase 1: copy x -> out, gather embeddings -> LDS (bf16) ----
  #pragma unroll
  for (int i = tid; i < 1024; i += 256) {
    const int s = i >> 8, c = i & 255;
    out[(n0 + s) * OUTW_ + c] = x[(n0 + s) * 256 + c];
  }
  {
    const int row = tid >> 2;                // 0..63 token row
    const int seg = tid & 3;                 // 4 threads/row, 192 cols each
    const int s = row >> 4, l = row & 15;
    const int name = names[(n0 + s) * 16 + l];
    const float* src = wte + name * D_ + seg * 192;
    __bf16* dst = emb + row * D_ + seg * 192;
    #pragma unroll
    for (int j = 0; j < 48; ++j) {
      float4 f = *(const float4*)(src + j * 4);
      v4bf b;
      b[0] = (__bf16)f.x; b[1] = (__bf16)f.y;
      b[2] = (__bf16)f.z; b[3] = (__bf16)f.w;
      *(v4bf*)(dst + j * 4) = b;
    }
  }
  __syncthreads();

  // ================= per-head pipeline =================
  #pragma unroll 1
  for (int h = 0; h < 4; ++h) {
    const int hd0 = h * DH_;                 // head column offset in D

    // ---- Phase 2h: QKV slice GEMM for this head.
    // 36 N-tiles (12 Q + 12 K + 12 V) over 8 waves; 4 M-tile accumulators
    // interleaved so one B fragment feeds 4 back-to-back WMMAs.
    #pragma unroll 1
    for (int i = 0; i < 5; ++i) {
      const int nt = wid + (i << 3);
      if (nt >= 36) break;
      const int g = nt / 12;                 // 0=Q, 1=K, 2=V
      const int t = nt % 12;                 // tile within the head slice
      const int wrow = g * D_ + hd0 + t * 16;  // W1 row base (output channel)
      const float bias = b1[wrow + (lane & 15)];
      v8f acc[4];
      #pragma unroll
      for (int mt = 0; mt < 4; ++mt)
        #pragma unroll
        for (int r = 0; r < 8; ++r) acc[mt][r] = bias;
      for (int kc = 0; kc < 24; ++kc) {      // K = 768
        v16bf b = ld_frag(w1, wrow, D_, kc * 32);
        #pragma unroll
        for (int mt = 0; mt < 4; ++mt) {
          v16bf a = ld_frag(emb, mt * 16, D_, kc * 32);
          acc[mt] = wmma_bf16(a, b, acc[mt]);
        }
      }
      const int el = t * 16 + (lane & 15);   // 0..191 within head slice
      #pragma unroll
      for (int mt = 0; mt < 4; ++mt) {
        #pragma unroll
        for (int r = 0; r < 8; ++r) {
          const int tok = mrow + r;          // token within sample
          const int row = mt * 16 + tok;     // 0..63
          const __bf16 v = (__bf16)acc[mt][r];
          if (g == 0)      Qh[row * DH_ + el] = v;
          else if (g == 1) Kh[row * DH_ + el] = v;
          else             Vth[(mt * DH_ + el) * 16 + tok] = v;
        }
      }
    }
    __syncthreads();

    // ---- Phase 3h-a: scores + softmax; waves 0-3, one sample each ----
    if (wid < 4) {
      const int s = wid;
      v8f c;
      #pragma unroll
      for (int r = 0; r < 8; ++r) c[r] = 0.f;
      #pragma unroll
      for (int kc = 0; kc < 6; ++kc) {       // DH = 192
        v16bf a = ld_frag(Qh + s * 16 * DH_, 0, DH_, kc * 32);
        v16bf b = ld_frag(Kh + s * 16 * DH_, 0, DH_, kc * 32);
        c = wmma_bf16(a, b, c);
      }
      float*  msc = sc + s * 256;
      __bf16* maa = aa + s * 256;
      const float scale = 0.07216878364870323f;   // 1/sqrt(192)
      #pragma unroll
      for (int r = 0; r < 8; ++r)
        msc[(mrow + r) * 16 + (lane & 15)] = c[r] * scale;
      if (lane < 16) {                       // one softmax row per lane
        float mx = -3.4e38f;
        #pragma unroll
        for (int j = 0; j < 16; ++j) mx = fmaxf(mx, msc[lane * 16 + j]);
        float ex[16], sum = 0.f;
        #pragma unroll
        for (int j = 0; j < 16; ++j) {
          ex[j] = __expf(msc[lane * 16 + j] - mx);
          sum += ex[j];
        }
        const float inv = 1.f / sum;
        #pragma unroll
        for (int j = 0; j < 16; ++j)
          maa[lane * 16 + j] = (__bf16)(ex[j] * inv);
      }
    }
    __syncthreads();

    // ---- Phase 3h-b: ctx = attn @ V. 48 jobs (4 samples x 12 d-tiles). ----
    #pragma unroll 1
    for (int j = 0; j < 6; ++j) {
      const int jid = wid + (j << 3);        // 0..47
      const int s  = jid & 3;
      const int dt = jid >> 2;               // 0..11
      v16bf a = ld_frag_k16(aa + s * 256, 0, 16);
      v16bf b = ld_frag_k16(Vth + s * DH_ * 16, dt * 16, 16);
      v8f cc;
      #pragma unroll
      for (int r = 0; r < 8; ++r) cc[r] = 0.f;
      cc = wmma_bf16(a, b, cc);
      #pragma unroll
      for (int r = 0; r < 8; ++r)
        ctx[(s * 16 + mrow + r) * D_ + hd0 + dt * 16 + (lane & 15)] =
            (__bf16)cc[r];
    }
    __syncthreads();                         // before next head reuses Qh/Kh/Vth
  }

  // ---- Phase 4: out-proj GEMM; pooling over L folded into epilogue ----
  #pragma unroll 1
  for (int i = 0; i < 6; ++i) {              // 48 N-tiles / 8 waves
    const int nt = wid + (i << 3);
    const int e  = nt * 16 + (lane & 15);
    v8f acc[4];
    #pragma unroll
    for (int mt = 0; mt < 4; ++mt)
      #pragma unroll
      for (int r = 0; r < 8; ++r) acc[mt][r] = 0.f;
    for (int kc = 0; kc < 24; ++kc) {
      v16bf b = ld_frag(w2, nt * 16, D_, kc * 32);
      #pragma unroll
      for (int mt = 0; mt < 4; ++mt) {
        v16bf a = ld_frag(ctx, mt * 16, D_, kc * 32);
        acc[mt] = wmma_bf16(a, b, acc[mt]);
      }
    }
    #pragma unroll
    for (int mt = 0; mt < 4; ++mt) {
      // pooled[e] = sum over the 16 token rows of this tile
      float p = acc[mt][0] + acc[mt][1] + acc[mt][2] + acc[mt][3] +
                acc[mt][4] + acc[mt][5] + acc[mt][6] + acc[mt][7];
      p += __shfl_xor(p, 16);                // rows 0-7 half + rows 8-15 half
      if (lane < 16)
        out[(n0 + mt) * OUTW_ + 256 + e] = p + 16.f * b2[e];
    }
  }
}

// ---------------------------------------------------------------------------
extern "C" void kernel_launch(void* const* d_in, const int* in_sizes, int n_in,
                              void* d_out, int out_size, void* d_ws,
                              size_t ws_size, hipStream_t stream) {
  const float* x     = (const float*)d_in[0];
  const int*   names = (const int*)d_in[1];
  const float* wte   = (const float*)d_in[2];
  const float* w1f   = (const float*)d_in[3];
  const float* b1    = (const float*)d_in[4];
  const float* w2f   = (const float*)d_in[5];
  const float* b2    = (const float*)d_in[6];
  float* out = (float*)d_out;

  unsigned short* w1b = (unsigned short*)d_ws;       // NW1_ bf16
  unsigned short* w2b = w1b + NW1_;                  // NW2_ bf16 (~4.5 MB total)

  cvt_weights_kernel<<<2304, 256, 0, stream>>>(w1f, w2f, w1b, w2b);
  encoder_kernel<<<2048, 256, SMEM_BYTES, stream>>>(x, names, wte, w1b, b1,
                                                    w2b, b2, out);
}
